// MemeGeneratorM2_87625922773807
// MI455X (gfx1250) — matrix operand order repaired and checked by reading, not compiled, not run on hardware
//
#include <hip/hip_runtime.h>

typedef __attribute__((ext_vector_type(16))) __bf16 v16bf;
typedef __attribute__((ext_vector_type(8)))  float  v8f;

#define B_    128
#define T_    512
#define V_    128
#define E_    128
#define H_    1024

// ---- workspace layout (bytes) ----
#define OFF_TOK   0u          // int[65536]
#define OFF_P     262144u     // float[16384]
#define OFF_Q     327680u     // float[16384]
#define OFF_GIMG  393216u     // float[524288]   Gimg[b][4096]
#define OFF_GTOK  2490368u    // float[524288]   Gtok[v][4096]
#define OFF_C     4587520u    // float[131072]   c[b][1024]
#define OFF_BAR   5111808u    // unsigned[64]
#define OFF_PWHH  5112064u    // bf16[4194304]   packed W_hh B-fragments
#define OFF_PFC   13500672u   // bf16[131072]    packed W_fc B-fragments
#define OFF_HS    13762816u   // bf16[513*131072] h history (slice 0 = h0 = 0)

union AF { v16bf v; unsigned u[8]; };

__device__ __forceinline__ float sigm_(float x) { return 1.0f / (1.0f + __expf(-x)); }
__device__ __forceinline__ float tanh_(float x) { float e = __expf(2.0f * x); return 1.0f - 2.0f / (e + 1.0f); }

__device__ __forceinline__ v8f wmma_bf16(v16bf a, v16bf b, v8f c) {
    return __builtin_amdgcn_wmma_f32_16x16x32_bf16(false, a, false, b, (short)0, c, false, false);
}

// async global->LDS copy of one 16B packet (CDNA5 TDM-adjacent async path, ASYNCcnt)
__device__ __forceinline__ void async_copy_b128(unsigned lds_off, const void* gaddr) {
    asm volatile("global_load_async_to_lds_b128 %0, %1, off"
                 :: "v"(lds_off), "v"(gaddr) : "memory");
}
__device__ __forceinline__ void wait_asynccnt0() {
    asm volatile("s_wait_asynccnt 0" ::: "memory");
}

// -------- init: teacher-forced token stream, zero h0/c/barrier --------
__global__ void k_init(const int* __restrict__ x, const int* __restrict__ label,
                       int* __restrict__ tok, __bf16* __restrict__ hs0,
                       float* __restrict__ c, unsigned* __restrict__ bar) {
    int tid = blockIdx.x * blockDim.x + threadIdx.x;          // 0..131071
    if (tid < B_ * T_) {
        int b = tid >> 9, t = tid & 511;
        tok[tid] = (t == 0) ? x[b * T_] : label[b * T_ + t - 1];
    }
    if (tid < B_ * H_) { hs0[tid] = (__bf16)0.0f; c[tid] = 0.0f; }
    if (tid == 0) bar[0] = 0u;
}

// -------- P[b,e] = b_proj[e] + img_emb·W_proj[:, :32] ; Q[v,e] = emb_char·W_proj[:, 32:] --------
__global__ void k_pq(const int* __restrict__ input_img, const float* __restrict__ emb_img,
                     const float* __restrict__ emb_char, const float* __restrict__ W_proj,
                     const float* __restrict__ b_proj, float* __restrict__ P, float* __restrict__ Q) {
    int tid = blockIdx.x * blockDim.x + threadIdx.x;          // 0..32767
    int which = tid >> 14, idx = tid & 16383;
    int row = idx >> 7, e = idx & 127;
    if (which == 0) {
        float acc = b_proj[e];
        int ii = input_img[row];
        for (int cc = 0; cc < 32; ++cc) acc += emb_img[ii * 32 + cc] * W_proj[e * 160 + cc];
        P[row * 128 + e] = acc;
    } else {
        float acc = 0.0f;
        for (int cc = 0; cc < 128; ++cc) acc += emb_char[row * 128 + cc] * W_proj[e * 160 + 32 + cc];
        Q[row * 128 + e] = acc;
    }
}

// -------- Gimg[b,g] = b_ih+b_hh + P@W_ihT ; Gtok[v,g] = Q@W_ihT --------
__global__ void k_gconst(const float* __restrict__ P, const float* __restrict__ Q,
                         const float* __restrict__ W_ih, const float* __restrict__ b_ih,
                         const float* __restrict__ b_hh, float* __restrict__ Gimg,
                         float* __restrict__ Gtok) {
    int tid = blockIdx.x * blockDim.x + threadIdx.x;          // 0..1048575
    int which = tid >> 19, idx = tid & 524287;
    int row = idx >> 12, g = idx & 4095;
    const float* src = which ? (Q + row * 128) : (P + row * 128);
    const float* wr  = W_ih + (size_t)g * 128;
    float acc = which ? 0.0f : (b_ih[g] + b_hh[g]);
    for (int e = 0; e < 128; ++e) acc += src[e] * wr[e];
    (which ? Gtok : Gimg)[(size_t)row * 4096 + g] = acc;
}

// -------- pack W_hh / W_fc into WMMA B-fragment order (lane=n, elem=k) --------
__global__ void k_pack(const float* __restrict__ W_hh, const float* __restrict__ W_fc,
                       __bf16* __restrict__ pwhh, __bf16* __restrict__ pfc) {
    long tid = (long)blockIdx.x * blockDim.x + threadIdx.x;   // 0..4325375
    if (tid < 4194304L) {
        int e = (int)(tid & 15), lane = (int)((tid >> 4) & 31);
        int ks = (int)((tid >> 9) & 31), nT = (int)(tid >> 14);      // nT 0..255
        int n = nT * 16 + (lane & 15);
        int k = ks * 32 + (e >> 3) * 16 + (lane >> 4) * 8 + (e & 7);
        pwhh[tid] = (__bf16)W_hh[(size_t)n * 1024 + k];
    } else if (tid < 4194304L + 131072L) {
        long o = tid - 4194304L;
        int e = (int)(o & 15), lane = (int)((o >> 4) & 31);
        int ks = (int)((o >> 9) & 31), vT = (int)(o >> 14);          // vT 0..7
        int v = vT * 16 + (lane & 15);
        int k = ks * 32 + (e >> 3) * 16 + (lane >> 4) * 8 + (e & 7);
        pfc[o] = (__bf16)W_fc[(size_t)v * 1024 + k];
    }
}

// -------- persistent LSTM: 512 steps, 32 blocks x 512 thr, device-wide barrier --------
__global__ void __launch_bounds__(512, 1)
k_lstm(const __bf16* __restrict__ pwhh, const float* __restrict__ Gimg,
       const float* __restrict__ Gtok, const int* __restrict__ tok,
       float* __restrict__ c, __bf16* __restrict__ hsAll, unsigned* __restrict__ bar) {
    __shared__ __align__(16) __bf16 As[16 * 1024];            // 32 KB: 16 h-rows
    const int tid  = threadIdx.x;
    const int lane = tid & 31;
    const int wv   = tid >> 5;                                // 0..15
    const int W    = blockIdx.x * 16 + wv;                    // 0..511 wave-task
    const int mT   = W >> 6;                                  // 0..7   (batch tile)
    const int ht   = W & 63;                                  // 0..63  (hidden tile)
    const int m0   = mT * 16;
    const int hi   = lane >> 4, nl = lane & 15;
    const int hu   = ht * 16 + nl;                            // hidden unit for this lane
    const v16bf* PB = (const v16bf*)pwhh;
    const size_t bb0 = (size_t)(0 * 64 + ht) * 1024;
    const size_t bb1 = (size_t)(1 * 64 + ht) * 1024;
    const size_t bb2 = (size_t)(2 * 64 + ht) * 1024;
    const size_t bb3 = (size_t)(3 * 64 + ht) * 1024;
    const unsigned ldsbase = (unsigned)(uintptr_t)(void*)&As[0];

    // Gimg is t-invariant: cache per-reg contributions
    float gim[8][4];
    #pragma unroll
    for (int r = 0; r < 8; ++r) {
        int b = m0 + r + 8 * hi;
        #pragma unroll
        for (int j = 0; j < 4; ++j) gim[r][j] = Gimg[(size_t)b * 4096 + j * 1024 + hu];
    }
    const unsigned* arow = (const unsigned*)As + nl * 512;    // 1024 bf16 = 512 dwords/row

    for (int t = 0; t < T_; ++t) {
        const __bf16* hprev = hsAll + (size_t)t * 131072;
        __bf16* hnext       = hsAll + (size_t)(t + 1) * 131072;
        {   // async-stage 16 rows of h (bf16, 32KB) straight into LDS (no VGPR round trip)
            const __bf16* gsrc = hprev + (size_t)m0 * 1024;
            for (int i = tid; i < 2048; i += 512)
                async_copy_b128(ldsbase + (unsigned)i * 16u, gsrc + (size_t)i * 8);
            wait_asynccnt0();
        }
        __syncthreads();

        v8f a0 = {0,0,0,0,0,0,0,0}, a1 = a0, a2 = a0, a3 = a0;
        for (int ks = 0; ks < 32; ++ks) {
            AF af;                                            // A frag per ISA 16-bit swizzle
            #pragma unroll
            for (int p = 0; p < 4; ++p) af.u[p] = arow[ks * 16 + hi * 4 + p];
            #pragma unroll
            for (int p = 4; p < 8; ++p) af.u[p] = arow[ks * 16 + hi * 4 + p + 4];
            v16bf b0v = PB[bb0 + ks * 32 + lane];
            v16bf b1v = PB[bb1 + ks * 32 + lane];
            v16bf b2v = PB[bb2 + ks * 32 + lane];
            v16bf b3v = PB[bb3 + ks * 32 + lane];
            a0 = wmma_bf16(af.v, b0v, a0);
            a1 = wmma_bf16(af.v, b1v, a1);
            a2 = wmma_bf16(af.v, b2v, a2);
            a3 = wmma_bf16(af.v, b3v, a3);
        }

        // fused LSTM cell (this wave owns i,f,g,o of 16 hidden units x 16 batch rows)
        #pragma unroll
        for (int r = 0; r < 8; ++r) {
            int b  = m0 + r + 8 * hi;
            int tk = tok[b * T_ + t];
            const float* gt = Gtok + (size_t)tk * 4096 + hu;
            float gi = a0[r] + gim[r][0] + gt[0];
            float gf = a1[r] + gim[r][1] + gt[1024];
            float gg = a2[r] + gim[r][2] + gt[2048];
            float go = a3[r] + gim[r][3] + gt[3072];
            int ci = b * 1024 + hu;
            float cn = sigm_(gf) * c[ci] + sigm_(gi) * tanh_(gg);
            c[ci] = cn;
            hnext[ci] = (__bf16)(sigm_(go) * tanh_(cn));
        }

        // device-wide barrier (monotonic counter, no reset needed)
        __threadfence();
        __syncthreads();
        if (tid == 0) {
            atomicAdd(bar, 1u);
            unsigned tgt = (unsigned)gridDim.x * (unsigned)(t + 1);
            while (__hip_atomic_load(bar, __ATOMIC_ACQUIRE, __HIP_MEMORY_SCOPE_AGENT) < tgt)
                __builtin_amdgcn_s_sleep(1);
        }
        __syncthreads();
        __threadfence();
    }
}

// -------- logits: [65536 x 128] = hs(bf16) @ W_fcT + b_fc, one big WMMA GEMM --------
__global__ void __launch_bounds__(256, 1)
k_logits(const __bf16* __restrict__ hsAll, const __bf16* __restrict__ pfc,
         const float* __restrict__ b_fc, float* __restrict__ out) {
    __shared__ __align__(16) __bf16 As[16 * 1024];
    const int tid = threadIdx.x;
    const int lane = tid & 31;
    const int vT = tid >> 5;                                  // 0..7
    const int m0 = blockIdx.x * 16;                           // rho = t*128 + b
    const int hi = lane >> 4, nl = lane & 15;
    {
        const __bf16* gsrc = hsAll + 131072 + (size_t)m0 * 1024;
        const unsigned ldsbase = (unsigned)(uintptr_t)(void*)&As[0];
        for (int i = tid; i < 2048; i += 256)
            async_copy_b128(ldsbase + (unsigned)i * 16u, gsrc + (size_t)i * 8);
        wait_asynccnt0();
    }
    __syncthreads();
    const unsigned* arow = (const unsigned*)As + nl * 512;
    const v16bf* PF = (const v16bf*)pfc;
    const size_t bb = (size_t)vT * 1024;
    v8f acc = {0,0,0,0,0,0,0,0};
    for (int ks = 0; ks < 32; ++ks) {
        AF af;
        #pragma unroll
        for (int p = 0; p < 4; ++p) af.u[p] = arow[ks * 16 + hi * 4 + p];
        #pragma unroll
        for (int p = 4; p < 8; ++p) af.u[p] = arow[ks * 16 + hi * 4 + p + 4];
        v16bf bv = PF[bb + ks * 32 + lane];
        acc = wmma_bf16(af.v, bv, acc);
    }
    int v = vT * 16 + nl;
    float bias = b_fc[v];
    #pragma unroll
    for (int r = 0; r < 8; ++r) {
        int rho = m0 + r + 8 * hi;
        int b = rho & 127, t = rho >> 7;
        out[((size_t)b * T_ + t) * V_ + v] = acc[r] + bias;
    }
}

// -------- argmax over V=128 (first-max semantics like jnp.argmax) --------
__global__ void k_argmax(const float* __restrict__ logits, float* __restrict__ preds) {
    int row = blockIdx.x * blockDim.x + threadIdx.x;          // 0..65535 (= b*512+t)
    const float* p = logits + (size_t)row * 128;
    float best = p[0]; int bi = 0;
    for (int j = 1; j < 128; ++j) { float val = p[j]; if (val > best) { best = val; bi = j; } }
    preds[row] = (float)bi;
}

extern "C" void kernel_launch(void* const* d_in, const int* in_sizes, int n_in,
                              void* d_out, int out_size, void* d_ws, size_t ws_size,
                              hipStream_t stream) {
    const int*   input_img = (const int*)d_in[0];
    const int*   x         = (const int*)d_in[1];
    const int*   label     = (const int*)d_in[2];
    const float* emb_img   = (const float*)d_in[3];
    const float* emb_char  = (const float*)d_in[4];
    const float* W_proj    = (const float*)d_in[5];
    const float* b_proj    = (const float*)d_in[6];
    const float* W_ih      = (const float*)d_in[7];
    const float* W_hh      = (const float*)d_in[8];
    const float* b_ih      = (const float*)d_in[9];
    const float* b_hh      = (const float*)d_in[10];
    const float* W_fc      = (const float*)d_in[11];
    const float* b_fc      = (const float*)d_in[12];

    char* ws = (char*)d_ws;
    int*      tok   = (int*)(ws + OFF_TOK);
    float*    P     = (float*)(ws + OFF_P);
    float*    Q     = (float*)(ws + OFF_Q);
    float*    Gimg  = (float*)(ws + OFF_GIMG);
    float*    Gtok  = (float*)(ws + OFF_GTOK);
    float*    c     = (float*)(ws + OFF_C);
    unsigned* bar   = (unsigned*)(ws + OFF_BAR);
    __bf16*   pwhh  = (__bf16*)(ws + OFF_PWHH);
    __bf16*   pfc   = (__bf16*)(ws + OFF_PFC);
    __bf16*   hsAll = (__bf16*)(ws + OFF_HS);
    float*    out   = (float*)d_out;

    k_init  <<<512, 256, 0, stream>>>(x, label, tok, hsAll, c, bar);
    k_pq    <<<128, 256, 0, stream>>>(input_img, emb_img, emb_char, W_proj, b_proj, P, Q);
    k_pack  <<<16896, 256, 0, stream>>>(W_hh, W_fc, pwhh, pfc);
    k_gconst<<<4096, 256, 0, stream>>>(P, Q, W_ih, b_ih, b_hh, Gimg, Gtok);
    k_lstm  <<<32, 512, 0, stream>>>(pwhh, Gimg, Gtok, tok, c, hsAll, bar);
    k_logits<<<4096, 256, 0, stream>>>(hsAll, pfc, b_fc, out);
    k_argmax<<<256, 256, 0, stream>>>(out, out + 8388608);
}
